// PositionAwareAttention_9706626089139
// MI455X (gfx1250) — matrix operand order repaired and checked by reading, not compiled, not run on hardware
//
#include <hip/hip_runtime.h>
#include <hip/hip_bf16.h>

#define S_ 8192
#define B_ 64
#define D_ 512
#define A_ 256

#define SCHUNK 64          // s values per workgroup in score kernel
#define XS_STRIDE 520      // bf16 elems per LDS row: 512 + 8 pad -> 260 dwords (stride%64==4 banks)
#define SSPLIT 32          // split-S factor for weighted sum

typedef __attribute__((ext_vector_type(16))) __bf16 v16bf;
typedef __attribute__((ext_vector_type(8)))  __bf16 v8bf;
typedef __attribute__((ext_vector_type(8)))  float  v8f;
typedef __attribute__((ext_vector_type(4)))  float  v4f;

// ---------------- kernel 1: Wx fp32 -> bf16 copy (L2-resident operand) ----------------
__global__ void k_cvt_wx(const float* __restrict__ Wx, __bf16* __restrict__ Wxb) {
    int i = blockIdx.x * blockDim.x + threadIdx.x;
    if (i < A_ * D_) Wxb[i] = (__bf16)Wx[i];
}

// ---------------- kernel 2: whc[b,a] = dot(h[b], Wh[a]) + bx[a] ----------------
__global__ void k_whc(const float* __restrict__ h, const float* __restrict__ Wh,
                      const float* __restrict__ bx, float* __restrict__ whc) {
    int b = blockIdx.x;
    int a = threadIdx.x;
    const float* hr = h + b * D_;
    const float* wr = Wh + a * D_;
    float acc = 0.f;
    for (int d = 0; d < D_; d += 4) {
        v4f hv = *(const v4f*)(hr + d);
        v4f wv = *(const v4f*)(wr + d);
        acc += hv.x * wv.x + hv.y * wv.y + hv.z * wv.z + hv.w * wv.w;
    }
    whc[b * A_ + a] = acc + bx[a];
}

// ---------------- kernel 3: score GEMM via bf16 WMMA ----------------
// grid (S/SCHUNK, B), 256 threads = 8 waves.
// wave w computes pre[32w..32w+31][s0..s0+63] = Wx * x^T, then tanh/dot(Wt) partials.
__global__ __launch_bounds__(256)
void k_score(const float* __restrict__ x, const __bf16* __restrict__ Wxb,
             const float* __restrict__ whc, const float* __restrict__ Wt,
             const float* __restrict__ bt, float* __restrict__ scores) {
    extern __shared__ char smem[];
    __bf16* xs    = (__bf16*)smem;                                   // [SCHUNK][XS_STRIDE]
    float*  spart = (float*)(smem + SCHUNK * XS_STRIDE * 2);         // [8][SCHUNK]

    const int b   = blockIdx.y;
    const int s0  = blockIdx.x * SCHUNK;
    const int tid = threadIdx.x;

    // stage x tile (64 x 512 fp32) -> LDS bf16, padded rows
    for (int i = tid; i < SCHUNK * (D_ / 4); i += 256) {
        int r = i >> 7;            // / (512/4)
        int c = (i & 127) << 2;
        v4f v = *(const v4f*)(x + (size_t)(s0 + r) * (B_ * D_) + (size_t)b * D_ + c);
        __bf16* dst = xs + r * XS_STRIDE + c;
        dst[0] = (__bf16)v.x; dst[1] = (__bf16)v.y;
        dst[2] = (__bf16)v.z; dst[3] = (__bf16)v.w;
    }
    __syncthreads();

    const int w    = tid >> 5;
    const int lane = tid & 31;
    const int l16  = lane & 15;
    const int half = lane >> 4;

    v8f acc[2][4] = {};

    const __bf16* wxrow = Wxb + (size_t)(32 * w + l16) * D_;
    for (int kk = 0; kk < 16; ++kk) {
        const int k0 = kk * 32;
        // A fragments: ISA layout -> lane half picks K {0..7,16..23} vs {8..15,24..31}
        v16bf afrag[2];
#pragma unroll
        for (int mt = 0; mt < 2; ++mt) {
            const __bf16* p = wxrow + mt * 16 * D_ + k0 + 8 * half;
            v8bf lo = *(const v8bf*)p;
            v8bf hi = *(const v8bf*)(p + 16);
            afrag[mt] = __builtin_shufflevector(lo, hi, 0,1,2,3,4,5,6,7,8,9,10,11,12,13,14,15);
        }
        // B fragments from LDS: lane half picks K {0..7,16..23} vs {8..15,24..31}, N = l16
#pragma unroll
        for (int nt = 0; nt < 4; ++nt) {
            const __bf16* q = xs + (nt * 16 + l16) * XS_STRIDE + k0 + 8 * half;
            v8bf lo = *(const v8bf*)q;
            v8bf hi = *(const v8bf*)(q + 16);
            v16bf bfrag = __builtin_shufflevector(lo, hi, 0,1,2,3,4,5,6,7,8,9,10,11,12,13,14,15);
#pragma unroll
            for (int mt = 0; mt < 2; ++mt) {
                acc[mt][nt] = __builtin_amdgcn_wmma_f32_16x16x32_bf16(
                    false, afrag[mt], false, bfrag, (short)0, acc[mt][nt], false, false);
            }
        }
    }

    // score partials: element j of acc tile -> a = 32w + 16mt + 8*half + j, s = s0 + 16nt + l16
    const int abase0 = 32 * w + 8 * half;
#pragma unroll
    for (int nt = 0; nt < 4; ++nt) {
        float p = 0.f;
#pragma unroll
        for (int mt = 0; mt < 2; ++mt) {
            const int abase = abase0 + 16 * mt;
#pragma unroll
            for (int j = 0; j < 8; ++j) {
                const int a = abase + j;
                float t = tanhf(acc[mt][nt][j] + whc[b * A_ + a]);
                p += Wt[a] * t;
            }
        }
        p += __shfl_down(p, 16, 32);          // combine lane halves (same s, complementary a)
        if (half == 0) spart[w * SCHUNK + nt * 16 + l16] = p;
    }
    __syncthreads();

    if (tid < SCHUNK) {
        float sum = 0.f;
#pragma unroll
        for (int w2 = 0; w2 < 8; ++w2) sum += spart[w2 * SCHUNK + tid];
        scores[(size_t)b * S_ + s0 + tid] = sum + bt[0];
    }
}

// ---------------- kernel 4: softmax over S per batch (in place) ----------------
__global__ __launch_bounds__(256)
void k_softmax(float* __restrict__ sc) {
    __shared__ float red[256];
    const int b = blockIdx.x, tid = threadIdx.x;
    float* row = sc + (size_t)b * S_;

    float m = -3.402823e38f;
    for (int i = tid; i < S_; i += 256) m = fmaxf(m, row[i]);
    red[tid] = m; __syncthreads();
    for (int o = 128; o > 0; o >>= 1) {
        if (tid < o) red[tid] = fmaxf(red[tid], red[tid + o]);
        __syncthreads();
    }
    m = red[0]; __syncthreads();

    float l = 0.f;
    for (int i = tid; i < S_; i += 256) l += expf(row[i] - m);
    red[tid] = l; __syncthreads();
    for (int o = 128; o > 0; o >>= 1) {
        if (tid < o) red[tid] += red[tid + o];
        __syncthreads();
    }
    const float inv = 1.f / red[0];

    for (int i = tid; i < S_; i += 256) row[i] = expf(row[i] - m) * inv;
}

// ---------------- kernel 5: split-S weighted sum partials (deterministic) ----------------
__global__ __launch_bounds__(512)
void k_wsum_part(const float* __restrict__ x, const float* __restrict__ attn,
                 float* __restrict__ part) {
    const int b = blockIdx.y, sp = blockIdx.x, d = threadIdx.x;
    const int s0 = sp * (S_ / SSPLIT);
    float acc = 0.f;
    for (int s = s0; s < s0 + S_ / SSPLIT; ++s)
        acc += attn[(size_t)b * S_ + s] * x[(size_t)s * (B_ * D_) + (size_t)b * D_ + d];
    part[((size_t)b * SSPLIT + sp) * D_ + d] = acc;
}

// ---------------- kernel 6: reduce partials -> out (B, D) ----------------
__global__ __launch_bounds__(512)
void k_wsum_final(const float* __restrict__ part, float* __restrict__ out) {
    const int b = blockIdx.x, d = threadIdx.x;
    float acc = 0.f;
#pragma unroll
    for (int sp = 0; sp < SSPLIT; ++sp)
        acc += part[((size_t)b * SSPLIT + sp) * D_ + d];
    out[b * D_ + d] = acc;
}

extern "C" void kernel_launch(void* const* d_in, const int* in_sizes, int n_in,
                              void* d_out, int out_size, void* d_ws, size_t ws_size,
                              hipStream_t stream) {
    const float* x  = (const float*)d_in[0];
    const float* h  = (const float*)d_in[1];
    const float* Wx = (const float*)d_in[2];
    const float* bx = (const float*)d_in[3];
    const float* Wh = (const float*)d_in[4];
    const float* Wt = (const float*)d_in[5];
    const float* bt = (const float*)d_in[6];
    float* out = (float*)d_out;

    char* ws = (char*)d_ws;
    float*  whc    = (float*)ws;                                   // 64*256 f   = 64 KB
    __bf16* Wxb    = (__bf16*)(ws + 65536);                        // 256*512 bf = 256 KB
    float*  scores = (float*)(ws + 65536 + 262144);                // 64*8192 f  = 2 MB
    float*  part   = (float*)(ws + 65536 + 262144 + 2097152);      // 64*32*512 f= 4 MB

    k_cvt_wx<<<dim3((A_ * D_) / 256), dim3(256), 0, stream>>>(Wx, Wxb);
    k_whc<<<dim3(B_), dim3(A_), 0, stream>>>(h, Wh, bx, whc);

    const size_t smem = (size_t)SCHUNK * XS_STRIDE * sizeof(__bf16) + 8 * SCHUNK * sizeof(float);
    k_score<<<dim3(S_ / SCHUNK, B_), dim3(256), smem, stream>>>(x, Wxb, whc, Wt, bt, scores);

    k_softmax<<<dim3(B_), dim3(256), 0, stream>>>(scores);
    k_wsum_part<<<dim3(SSPLIT, B_), dim3(D_), 0, stream>>>(x, scores, part);
    k_wsum_final<<<dim3(B_), dim3(D_), 0, stream>>>(part, out);
}
